// k1_GNN_7842610283372
// MI455X (gfx1250) — matrix operand order repaired and checked by reading, not compiled, not run on hardware
//
#include <hip/hip_runtime.h>

// ---------------------------------------------------------------------------
// NNConv GNN forward for gfx1250 (MI455X).
//
// Dominant cost: per-layer GEMM [E,128] @ [128, mi*mo]  (134 GFLOP total).
// Reformulated as one GEMM [E, 128*mi] @ [128*mi, mo] whose A operand
// P[e,(i,k)] = h[e,k]*x_src[e,i] is built on the fly: cached h A-fragments
// row-scaled by x_src scalars via v_pk_mul_bf16, accumulated with
// v_wmma_f32_16x16x32_bf16 (f32 accumulators), b2 folded in as 2 extra
// K-steps, result scattered with global f32 atomics.
//
// M-tile = 32 edges per wave (two A-row sets sharing each B fragment) to
// halve B-fragment L2 traffic; w2 is pre-packed once per layer (~1MB, L2
// resident) into B-fragment order (one contiguous 32B load per lane).
// ---------------------------------------------------------------------------

typedef __attribute__((ext_vector_type(16))) __bf16 v16bf;
typedef __attribute__((ext_vector_type(8)))  float  v8f;

union FragBF { unsigned u[8]; v16bf v; };

__device__ __forceinline__ unsigned f2bf_rne(float f) {
  unsigned u = __float_as_uint(f);
  return (u + 0x7FFFu + ((u >> 16) & 1u)) >> 16;   // round-to-nearest-even bf16
}

__device__ __forceinline__ float elu1(float x) {
  return x > 0.0f ? x : (expf(x) - 1.0f);
}

// --------------------------- tiny helper kernels ---------------------------

__global__ void zero_f32_k(float* p, int n) {
  int t = blockIdx.x * blockDim.x + threadIdx.x;
  if (t < n) p[t] = 0.0f;
}

// h[e,k] = bf16(relu(edge_attr[e,:]@w1[:,k] + b1[k])), row-major [E,128] bf16.
__global__ void edge_mlp1_k(const float* __restrict__ ea, const float* __restrict__ w1,
                            const float* __restrict__ b1, unsigned short* __restrict__ hbf,
                            int E, int EA) {
  int t = blockIdx.x * blockDim.x + threadIdx.x;
  if (t >= E * 128) return;
  int e = t >> 7, k = t & 127;
  float a = b1[k];
  for (int j = 0; j < EA; ++j) a += ea[e * EA + j] * w1[j * 128 + k];
  a = a > 0.0f ? a : 0.0f;
  hbf[t] = (unsigned short)f2bf_rne(a);
}

// Pre-pack w2 (f32 [128, MI*MO]) + b2 into bf16 B-fragments.
// Fragment (s, ob) = 32x16 (K'xN) tile; storage: lane-major 32B per lane.
// B element (qlocal, n): lane = 16*((qlocal>>3)&1) + n, dword v=4*(qlocal>>4)+((qlocal&7)>>1),
// half = qlocal&1. K' index q = 32*s + qlocal -> i = q>>7, k = q&127 for main
// steps; the last 2 steps hold b2 rows (zero-padded past mi).
template <int MI, int MO>
__global__ void prepack_b_k(const float* __restrict__ w2, const float* __restrict__ b2,
                            unsigned* __restrict__ Bp) {
  constexpr int OB = MO / 16;
  constexpr int SMAIN = 4 * MI;          // 128*MI / 32
  constexpr int S = SMAIN + 2;           // + bias steps (cover i in [0,64))
  constexpr int CW = MI * MO;
  int t = blockIdx.x * blockDim.x + threadIdx.x;
  if (t >= S * OB * 256) return;         // one dword (2 bf16) per thread
  int dw = t & 255, frag = t >> 8;
  int ob = frag % OB, s = frag / OB;
  int lane = dw >> 3, v = dw & 7;
  int n = lane & 15;
  unsigned out = 0;
  for (int half = 0; half < 2; ++half) {
    int sl = v * 2 + half;
    int ql = (sl < 8) ? ((lane >> 4) * 8 + sl) : (16 + (lane >> 4) * 8 + (sl - 8));
    float val;
    if (s < SMAIN) {
      int q = 32 * s + ql;
      int i = q >> 7, k = q & 127;
      val = w2[(size_t)k * CW + i * MO + ob * 16 + n];
    } else {
      int i = 32 * (s - SMAIN) + ql;
      val = (i < MI) ? b2[i * MO + ob * 16 + n] : 0.0f;
    }
    out |= f2bf_rne(val) << (16 * half);
  }
  Bp[(size_t)frag * 256 + dw] = out;
}

// ----------------------- fused GEMM + per-edge scatter ----------------------
// One wave per 32-edge tile (two 16-row subtiles sharing each B fragment);
// all MO/16 o-blocks accumulated simultaneously. K' = 128*MI (+64 for bias).
template <int MI, int MO>
__global__ void __launch_bounds__(32)
nnconv_gemm_scatter_k(const unsigned short* __restrict__ hbf, const float* __restrict__ xin,
                      const unsigned* __restrict__ Bp, const int* __restrict__ src,
                      const int* __restrict__ dst, float* __restrict__ agg, int E) {
  constexpr int OB = MO / 16;
  constexpr int SMAIN = 4 * MI;
  __shared__ float xs[32 * MI];

  const int lane = threadIdx.x;
  const int e0 = blockIdx.x * 32;
  const int m = lane & 15;       // A row / D column index
  const int hi = lane >> 4;

  // Stage x_src tile [32][MI] into LDS (gather via src[]).
  for (int t = lane; t < 32 * MI; t += 32) {
    int mm = t / MI, ii = t - mm * MI;
    int er = e0 + mm; if (er >= E) er = E - 1;
    xs[t] = xin[(size_t)src[er] * MI + ii];
  }
  __syncthreads();   // single-wave WG: lowers to nop; LDS same-wave ordered

  // Destination node ids for the D rows this lane scatters (per subtile).
  int dstv[2][8];
#pragma unroll
  for (int sub = 0; sub < 2; ++sub)
#pragma unroll
    for (int r = 0; r < 8; ++r) {
      int er = e0 + sub * 16 + hi * 8 + r; if (er >= E) er = E - 1;
      dstv[sub][r] = dst[er];
    }

  // Cache the 4 h A-fragments (K = 0..127) per subtile for this lane's row.
  // A layout (16-bit 16x32): lane m, two contiguous 8-element chunks at
  // K = hi*8 and K = 16 + hi*8.
  unsigned hfr[2][4][8];
#pragma unroll
  for (int sub = 0; sub < 2; ++sub) {
    int erA = e0 + sub * 16 + m; if (erA >= E) erA = E - 1;
    const unsigned short* rowp = hbf + (size_t)erA * 128;
#pragma unroll
    for (int f = 0; f < 4; ++f) {
      uint4 a = *(const uint4*)(rowp + f * 32 + hi * 8);
      uint4 b = *(const uint4*)(rowp + f * 32 + hi * 8 + 16);
      hfr[sub][f][0] = a.x; hfr[sub][f][1] = a.y; hfr[sub][f][2] = a.z; hfr[sub][f][3] = a.w;
      hfr[sub][f][4] = b.x; hfr[sub][f][5] = b.y; hfr[sub][f][6] = b.z; hfr[sub][f][7] = b.w;
    }
  }

  v8f acc[2][OB] = {};

  // Main K' loop: i selects the x_src scalar, kk the cached h fragment.
  for (int i = 0; i < MI; ++i) {
    unsigned xsb[2];
#pragma unroll
    for (int sub = 0; sub < 2; ++sub) {
      unsigned bb = f2bf_rne(xs[(sub * 16 + m) * MI + i]);
      xsb[sub] = bb | (bb << 16);
    }
#pragma unroll
    for (int kk = 0; kk < 4; ++kk) {
      FragBF af[2];
#pragma unroll
      for (int sub = 0; sub < 2; ++sub)
#pragma unroll
        for (int d = 0; d < 8; ++d) {
          unsigned r;
          asm("v_pk_mul_bf16 %0, %1, %2" : "=v"(r) : "v"(hfr[sub][kk][d]), "v"(xsb[sub]));
          af[sub].u[d] = r;
        }
      const int s = i * 4 + kk;
#pragma unroll
      for (int ob = 0; ob < OB; ++ob) {
        const unsigned* bp = Bp + (size_t)(s * OB + ob) * 256 + lane * 8;
        uint4 q0 = *(const uint4*)(bp);
        uint4 q1 = *(const uint4*)(bp + 4);
        FragBF bfr;
        bfr.u[0] = q0.x; bfr.u[1] = q0.y; bfr.u[2] = q0.z; bfr.u[3] = q0.w;
        bfr.u[4] = q1.x; bfr.u[5] = q1.y; bfr.u[6] = q1.z; bfr.u[7] = q1.w;
#pragma unroll
        for (int sub = 0; sub < 2; ++sub)
          acc[sub][ob] = __builtin_amdgcn_wmma_f32_16x16x32_bf16(
              false, af[sub].v, false, bfr.v, (short)0, acc[sub][ob], false, false);
      }
    }
  }

  // Bias K-steps: A = raw x_src (zero past MI, matching zero-padded B rows).
#pragma unroll
  for (int sp = 0; sp < 2; ++sp) {
    FragBF af[2];
#pragma unroll
    for (int sub = 0; sub < 2; ++sub)
#pragma unroll
      for (int d = 0; d < 8; ++d) {
        int j0 = 2 * d, j1 = 2 * d + 1;
        int K0 = (j0 < 8) ? (hi * 8 + j0) : (16 + hi * 8 + (j0 - 8));
        int K1 = (j1 < 8) ? (hi * 8 + j1) : (16 + hi * 8 + (j1 - 8));
        int i0 = 32 * sp + K0, i1 = 32 * sp + K1;
        float v0 = (i0 < MI) ? xs[(sub * 16 + m) * MI + i0] : 0.0f;
        float v1 = (i1 < MI) ? xs[(sub * 16 + m) * MI + i1] : 0.0f;
        af[sub].u[d] = f2bf_rne(v0) | (f2bf_rne(v1) << 16);
      }
    const int s = SMAIN + sp;
#pragma unroll
    for (int ob = 0; ob < OB; ++ob) {
      const unsigned* bp = Bp + (size_t)(s * OB + ob) * 256 + lane * 8;
      uint4 q0 = *(const uint4*)(bp);
      uint4 q1 = *(const uint4*)(bp + 4);
      FragBF bfr;
      bfr.u[0] = q0.x; bfr.u[1] = q0.y; bfr.u[2] = q0.z; bfr.u[3] = q0.w;
      bfr.u[4] = q1.x; bfr.u[5] = q1.y; bfr.u[6] = q1.z; bfr.u[7] = q1.w;
#pragma unroll
      for (int sub = 0; sub < 2; ++sub)
        acc[sub][ob] = __builtin_amdgcn_wmma_f32_16x16x32_bf16(
            false, af[sub].v, false, bfr.v, (short)0, acc[sub][ob], false, false);
    }
  }

  // Scatter D (layout: lane L, VGPR r -> row 8*hi + r, col = L&15).
#pragma unroll
  for (int sub = 0; sub < 2; ++sub)
#pragma unroll
    for (int ob = 0; ob < OB; ++ob)
#pragma unroll
      for (int r = 0; r < 8; ++r) {
        int er = e0 + sub * 16 + hi * 8 + r;
        if (er < E)
          atomicAdd(&agg[(size_t)dstv[sub][r] * MO + ob * 16 + m], acc[sub][ob][r]);
      }
}

// x_out = elu(x @ root + agg + bias)
template <int MI, int MO>
__global__ void combine_elu_k(const float* __restrict__ xin, const float* __restrict__ root,
                              const float* __restrict__ bias, const float* __restrict__ agg,
                              float* __restrict__ xout, int N) {
  int t = blockIdx.x * blockDim.x + threadIdx.x;
  if (t >= N * MO) return;
  int n = t / MO, o = t - n * MO;
  float a = bias[o] + agg[t];
  const float* xr = xin + (size_t)n * MI;
#pragma unroll 4
  for (int i = 0; i < MI; ++i) a += xr[i] * root[i * MO + o];
  xout[t] = elu1(a);
}

// segment-sum pooling (+ node counts)
__global__ void pool_scatter_k(const float* __restrict__ x, const int* __restrict__ batch,
                               float* __restrict__ sums, float* __restrict__ cnt, int N) {
  int t = blockIdx.x * blockDim.x + threadIdx.x;
  if (t >= N * 64) return;
  int n = t >> 6, c = t & 63;
  int g = batch[n];
  atomicAdd(&sums[(size_t)g * 64 + c], x[t]);
  if (c == 0) atomicAdd(&cnt[g], 1.0f);
}

// mean + fc1(64->32) elu + fc2(32->16) elu + fc3(16->1)
__global__ void head_k(const float* __restrict__ sums, const float* __restrict__ cnt,
                       const float* f1w, const float* f1b, const float* f2w, const float* f2b,
                       const float* f3w, const float* f3b, float* __restrict__ out, int G) {
  int g = blockIdx.x * blockDim.x + threadIdx.x;
  if (g >= G) return;
  float inv = 1.0f / fmaxf(cnt[g], 1.0f);
  float xr[64];
#pragma unroll
  for (int i = 0; i < 64; ++i) xr[i] = sums[(size_t)g * 64 + i] * inv;
  float h1[32];
  for (int o = 0; o < 32; ++o) {
    float s = f1b[o];
    for (int i = 0; i < 64; ++i) s += xr[i] * f1w[i * 32 + o];
    h1[o] = elu1(s);
  }
  float h2[16];
  for (int o = 0; o < 16; ++o) {
    float s = f2b[o];
    for (int i = 0; i < 32; ++i) s += h1[i] * f2w[i * 16 + o];
    h2[o] = elu1(s);
  }
  float s = f3b[0];
  for (int i = 0; i < 16; ++i) s += h2[i] * f3w[i];
  out[g] = s;
}

// --------------------------------- launcher ---------------------------------
// Param pointer order assumes recursive dict-insertion-order flattening of
// setup_inputs(): x, edge_attr, edge_index, batch, then per conv layer
// (w1,b1,w2,b2,root,bias) x3, then fc1_w,fc1_b,fc2_w,fc2_b,fc3_w,fc3_b.
extern "C" void kernel_launch(void* const* d_in, const int* in_sizes, int n_in,
                              void* d_out, int out_size, void* d_ws, size_t ws_size,
                              hipStream_t stream) {
  const int N = in_sizes[3];                 // batch has one entry per node
  const int E = in_sizes[2] / 2;             // edge_index is [2, E]
  const int EA = in_sizes[1] / E;            // edge_attr_dim
  const int G = out_size;

  const float* x_in  = (const float*)d_in[0];
  const float* ea    = (const float*)d_in[1];
  const int*   ei    = (const int*)d_in[2];
  const int*   batch = (const int*)d_in[3];
  const int* src = ei;
  const int* dst = ei + E;

  // Workspace carve-up (~42 MB total).
  char* ws = (char*)d_ws;
  size_t off = 0;
  auto carve = [&](size_t bytes) -> void* {
    void* p = ws + off;
    off = (off + bytes + 255) & ~(size_t)255;
    return p;
  };
  unsigned short* hbf = (unsigned short*)carve((size_t)E * 128 * 2);
  unsigned*       Bp  = (unsigned*)carve((size_t)(4 * 64 + 2) * 4 * 1024);  // max layer
  float* xb0  = (float*)carve((size_t)N * 64 * 4);
  float* xb1  = (float*)carve((size_t)N * 64 * 4);
  float* agg  = (float*)carve((size_t)N * 64 * 4);
  float* sums = (float*)carve((size_t)G * 64 * 4);
  float* cnt  = (float*)carve((size_t)G * 4);

  const int tileGrid = (E + 31) / 32;

#define RUN_LAYER(L, MI, MO, XCUR, XNEXT)                                              \
  do {                                                                                 \
    const float* w1   = (const float*)d_in[4 + 6 * (L) + 0];                           \
    const float* b1   = (const float*)d_in[4 + 6 * (L) + 1];                           \
    const float* w2   = (const float*)d_in[4 + 6 * (L) + 2];                           \
    const float* b2   = (const float*)d_in[4 + 6 * (L) + 3];                           \
    const float* root = (const float*)d_in[4 + 6 * (L) + 4];                           \
    const float* bias = (const float*)d_in[4 + 6 * (L) + 5];                           \
    int ndw = (4 * (MI) + 2) * ((MO) / 16) * 256;                                      \
    prepack_b_k<MI, MO><<<(ndw + 255) / 256, 256, 0, stream>>>(w2, b2, Bp);            \
    edge_mlp1_k<<<(E * 128 + 255) / 256, 256, 0, stream>>>(ea, w1, b1, hbf, E, EA);    \
    zero_f32_k<<<(N * (MO) + 255) / 256, 256, 0, stream>>>(agg, N * (MO));             \
    nnconv_gemm_scatter_k<MI, MO><<<tileGrid, 32, 0, stream>>>(hbf, (XCUR), Bp, src,   \
                                                               dst, agg, E);           \
    combine_elu_k<MI, MO><<<(N * (MO) + 255) / 256, 256, 0, stream>>>(                 \
        (XCUR), root, bias, agg, (XNEXT), N);                                          \
  } while (0)

  RUN_LAYER(0, 13, 32, x_in, xb0);
  RUN_LAYER(1, 32, 64, xb0, xb1);
  RUN_LAYER(2, 64, 64, xb1, xb0);
#undef RUN_LAYER

  zero_f32_k<<<(G * 64 + 255) / 256, 256, 0, stream>>>(sums, G * 64);
  zero_f32_k<<<(G + 255) / 256, 256, 0, stream>>>(cnt, G);
  pool_scatter_k<<<(N * 64 + 255) / 256, 256, 0, stream>>>(xb0, batch, sums, cnt, N);
  head_k<<<(G + 255) / 256, 256, 0, stream>>>(
      sums, cnt, (const float*)d_in[22], (const float*)d_in[23],
      (const float*)d_in[24], (const float*)d_in[25],
      (const float*)d_in[26], (const float*)d_in[27], (float*)d_out, G);
}
